// CrossAttnHead_33517924778259
// MI455X (gfx1250) — matrix operand order repaired and checked by reading, not compiled, not run on hardware
//
#include <hip/hip_runtime.h>

#define BB    16
#define HH    12
#define NTAR  256
#define NOBS  512
#define DD    128
#define NHEAD 4
#define DHEAD 32
#define BH    (BB*HH)        // 192
#define ROWS_KV (BB*NOBS)    // 8192
#define ROWS_Q  (BH*NTAR)    // 49152

typedef __attribute__((ext_vector_type(16))) __bf16         v16bf;
typedef __attribute__((ext_vector_type(8)))  float          v8f;
typedef __attribute__((ext_vector_type(16))) unsigned short v16u;
typedef __attribute__((ext_vector_type(8)))  unsigned short v8u;

static __device__ __forceinline__ unsigned short f2bf(float f) {
  union { __bf16 h; unsigned short u; } c;
  c.h = (__bf16)f;                       // hardware cvt (RNE)
  return c.u;
}
static __device__ __forceinline__ float bf2f(unsigned short u) {
  union { float f; unsigned int u; } c;
  c.u = ((unsigned int)u) << 16;
  return c.f;
}

static __device__ __forceinline__ v8f wmma_bf16(v16bf a, v16bf b, v8f c) {
  return __builtin_amdgcn_wmma_f32_16x16x32_bf16(false, a, false, b, (short)0, c, false, false);
}

// A-matrix 16x32 bf16 frag (ISA 7.12.2): per lane the two K-chunks are contiguous:
// K = half*8 + [0..7]  and  K = 16 + half*8 + [0..7]  -> two 16B loads
static __device__ __forceinline__ v16bf load_a_bf16(const unsigned short* p, int ld, int lane) {
  int row = lane & 15, half = lane >> 4;
  const unsigned short* r = p + row * ld + half * 8;
  v8u lo = *(const v8u*)(r);
  v8u hi = *(const v8u*)(r + 16);
  v16u u;
#pragma unroll
  for (int i = 0; i < 8; ++i) { u[i] = lo[i]; u[8 + i] = hi[i]; }
  return __builtin_bit_cast(v16bf, u);
}

// B-matrix 32x16 bf16 frag: element (k,n) = p[n*ld + k]; K = half*8 + [0..7] and K = 16 + half*8 + [0..7] -> two 16B loads
static __device__ __forceinline__ v16bf load_b_n(const unsigned short* p, int ld, int lane) {
  int n = lane & 15, half = lane >> 4;
  const unsigned short* r = p + n * ld + half * 8;
  v8u lo = *(const v8u*)(r);
  v8u hi = *(const v8u*)(r + 16);
  v16u u;
#pragma unroll
  for (int i = 0; i < 8; ++i) { u[i] = lo[i]; u[8 + i] = hi[i]; }
  return __builtin_bit_cast(v16bf, u);
}

static __device__ __forceinline__ v8f vzero8() {
  v8f z = {0.f, 0.f, 0.f, 0.f, 0.f, 0.f, 0.f, 0.f};
  return z;
}

// ---------------- prep kernels ----------------

__global__ void cvt_w_kernel(const float* Wk, const float* Wv, const float* W1,
                             unsigned short* Wkb, unsigned short* Wvb, unsigned short* W1b) {
  int i = blockIdx.x * 256 + threadIdx.x;
  if (i < DD * DD) { Wkb[i] = f2bf(Wk[i]); Wvb[i] = f2bf(Wv[i]); W1b[i] = f2bf(W1[i]); }
}

__global__ void cvt_x_kernel(const float* X, unsigned short* Xb) {
  int i = blockIdx.x * 256 + threadIdx.x;
  if (i < ROWS_KV * DD) Xb[i] = f2bf(X[i]);
}

// QA[bh][c] = sum_k A[bh][k] * Wq[c][k]   (first 128 input cols of Wq)
__global__ void qa_kernel(const float* A, const float* Wq, float* QA) {
  int i = blockIdx.x * 256 + threadIdx.x;
  if (i >= BH * DD) return;
  int r = i >> 7, c = i & 127;
  const float* a = A + r * 128;
  const float* w = Wq + c * 256;
  float s = 0.f;
#pragma unroll 8
  for (int k = 0; k < 128; ++k) s += a[k] * w[k];
  QA[i] = s;
}

// QP[t][c] = sum_k phi[t][k] * Wq[c][128+k] + bq[c]
__global__ void qphi_kernel(const float* phi, const float* Wq, const float* bq, float* QP) {
  int i = blockIdx.x * 256 + threadIdx.x;
  if (i >= NTAR * DD) return;
  int t = i >> 7, c = i & 127;
  const float* p = phi + t * 128;
  const float* w = Wq + c * 256 + 128;
  float s = bq[c];
#pragma unroll 8
  for (int k = 0; k < 128; ++k) s += p[k] * w[k];
  QP[i] = s;
}

// ---------------- K/V projection (WMMA) ----------------
// blockIdx.x in [0,1024): bit9 selects K vs V; 16-row output tile per wave.
// K row-major [b][o][d] bf16; V transposed [b][d][o] bf16.
// All 8 B-frags of a K-step loaded in one clause before the 8-WMMA burst.
__global__ void kv_proj_kernel(const unsigned short* Xb, const unsigned short* Wkb,
                               const unsigned short* Wvb, const float* bk, const float* bv,
                               unsigned short* Kb, unsigned short* Vtb) {
  int lane = threadIdx.x;
  int half = lane >> 4, nn = lane & 15;
  int tile = blockIdx.x & 511;
  int isV  = blockIdx.x >> 9;
  int row0 = tile * 16;

  const unsigned short* W = isV ? Wvb : Wkb;
  const float* bias = isV ? bv : bk;

  v8f acc[8];
#pragma unroll
  for (int i = 0; i < 8; ++i) acc[i] = vzero8();

#pragma unroll
  for (int ks = 0; ks < 4; ++ks) {
    v16bf a = load_a_bf16(Xb + row0 * DD + ks * 32, DD, lane);
    v16bf bfr[8];
#pragma unroll
    for (int nt = 0; nt < 8; ++nt)
      bfr[nt] = load_b_n(W + nt * 16 * DD + ks * 32, DD, lane);
#pragma unroll
    for (int nt = 0; nt < 8; ++nt)
      acc[nt] = wmma_bf16(a, bfr[nt], acc[nt]);
  }

  int b_  = tile >> 5;
  int ol0 = (tile & 31) * 16;
#pragma unroll
  for (int nt = 0; nt < 8; ++nt) {
    int n = nt * 16 + nn;
    float bb = bias[n];
#pragma unroll
    for (int r = 0; r < 8; ++r) {
      int m = r + 8 * half;
      unsigned short v = f2bf(acc[nt][r] + bb);
      if (isV) Vtb[b_ * (DD * NOBS) + n * NOBS + ol0 + m] = v;
      else     Kb[(row0 + m) * DD + n] = v;
    }
  }
}

// ---------------- fused attention + FFN ----------------
// One wave per 16-target-row tile of one (b,h): all 4 heads + output FFN
// (W1 GEMM + LayerNorm + ReLU + W2 GEMV) from LDS. 3072 waves.
// Q frags built in-register from QA + Qphi; scores bf16 in LDS; softmax
// normalization folded into PV accumulators; ctx never touches global.
__global__ void attn_ffn_kernel(const float* QA, const float* QP, const unsigned short* Kb,
                                const unsigned short* Vtb, const unsigned short* W1b,
                                const float* b1, const float* lng, const float* lnb,
                                const float* W2, const float* b2, float* out) {
  __shared__ __align__(32) unsigned short sp[16 * 512];    // 16 KB, reused per head
  __shared__ __align__(32) unsigned short ctxs[16 * DD];   // 4 KB bf16 ctx tile
  __shared__ __align__(16) float xs[16 * DD];              // 8 KB f32 post-W1
  __shared__ float invs[16];
  int lane  = threadIdx.x;
  int half  = lane >> 4, nn = lane & 15;
  int ttile = blockIdx.x & 15;
  int bh    = blockIdx.x >> 4;           // 0..191
  int b_    = bh / HH;
  int t0    = ttile * 16;

  const float scale = 0.17677669529663687f;  // 1/sqrt(32)

  for (int head = 0; head < NHEAD; ++head) {
    // --- Q A-frag: Q[t][c] = QA[bh][c] + QP[t][c], c = head*32 + k ---
    v16bf qa;
    {
      const float* qarow = QA + bh * DD + head * DHEAD + half * 8;
      const float* qprow = QP + (t0 + nn) * DD + head * DHEAD + half * 8;
      v8f a0 = *(const v8f*)(qarow)      + *(const v8f*)(qprow);
      v8f a1 = *(const v8f*)(qarow + 16) + *(const v8f*)(qprow + 16);
      v16u qu;
#pragma unroll
      for (int i = 0; i < 8; ++i) { qu[i] = f2bf(a0[i]); qu[8 + i] = f2bf(a1[i]); }
      qa = __builtin_bit_cast(v16bf, qu);
    }

    // --- scores = Q K^T: groups of 4 obs-tiles (4 clause-batched loads, 4 WMMAs) ---
    const unsigned short* kslice = Kb + b_ * NOBS * DD + head * DHEAD;
    for (int og = 0; og < 8; ++og) {
      if (og + 1 < 8)
        __builtin_prefetch(kslice + (og + 1) * 64 * DD, 0, 0);
      v16bf kb[4];
#pragma unroll
      for (int i = 0; i < 4; ++i)
        kb[i] = load_b_n(kslice + (og * 4 + i) * 16 * DD, DD, lane);
      v8f sg[4];
#pragma unroll
      for (int i = 0; i < 4; ++i)
        sg[i] = wmma_bf16(qa, kb[i], vzero8());
#pragma unroll
      for (int i = 0; i < 4; ++i)
#pragma unroll
        for (int r = 0; r < 8; ++r)
          sp[(r + 8 * half) * 512 + (og * 4 + i) * 16 + nn] = f2bf(sg[i][r] * scale);
    }

    // --- softmax (unnormalized exp): two lanes per row, shfl_xor(16) combine ---
    {
      unsigned short* rowp = &sp[nn * 512 + half * 256];
      float m = -3.4e38f;
      for (int j = 0; j < 256; ++j) m = fmaxf(m, bf2f(rowp[j]));
      m = fmaxf(m, __shfl_xor(m, 16, 32));
      float s = 0.f;
      for (int j = 0; j < 256; ++j) {
        float e = __expf(bf2f(rowp[j]) - m);
        rowp[j] = f2bf(e);
        s += e;
      }
      s += __shfl_xor(s, 16, 32);
      if (half == 0) invs[nn] = 1.f / s;
    }

    // --- ctx = P V (unnormalized): groups of 2 K-steps (6 loads, 4 WMMAs) ---
    v8f c0 = vzero8(), c1 = vzero8();
    const unsigned short* vbase = Vtb + b_ * (DD * NOBS) + head * DHEAD * NOBS;
    for (int kg = 0; kg < 8; ++kg) {
      v16bf pa0  = load_a_bf16(&sp[(2 * kg) * 32],     512, lane);
      v16bf pa1  = load_a_bf16(&sp[(2 * kg + 1) * 32], 512, lane);
      v16bf vb00 = load_b_n(vbase + (2 * kg) * 32,                 NOBS, lane);
      v16bf vb10 = load_b_n(vbase + 16 * NOBS + (2 * kg) * 32,     NOBS, lane);
      v16bf vb01 = load_b_n(vbase + (2 * kg + 1) * 32,             NOBS, lane);
      v16bf vb11 = load_b_n(vbase + 16 * NOBS + (2 * kg + 1) * 32, NOBS, lane);
      c0 = wmma_bf16(pa0, vb00, c0);
      c1 = wmma_bf16(pa0, vb10, c1);
      c0 = wmma_bf16(pa1, vb01, c0);
      c1 = wmma_bf16(pa1, vb11, c1);
    }

    // --- normalize and park ctx tile in LDS (bf16) ---
#pragma unroll
    for (int r = 0; r < 8; ++r) {
      int m = r + 8 * half;
      float inv = invs[m];
      ctxs[m * DD + head * DHEAD + nn]      = f2bf(c0[r] * inv);
      ctxs[m * DD + head * DHEAD + 16 + nn] = f2bf(c1[r] * inv);
    }
  }

  // --- FFN: x = ctx @ W1^T + b1 (WMMA from LDS, clause-batched B loads) ---
  v8f acc[8];
#pragma unroll
  for (int i = 0; i < 8; ++i) acc[i] = vzero8();
#pragma unroll
  for (int ks = 0; ks < 4; ++ks) {
    v16bf a = load_a_bf16(ctxs + ks * 32, DD, lane);
    v16bf bfr[8];
#pragma unroll
    for (int nt = 0; nt < 8; ++nt)
      bfr[nt] = load_b_n(W1b + nt * 16 * DD + ks * 32, DD, lane);
#pragma unroll
    for (int nt = 0; nt < 8; ++nt)
      acc[nt] = wmma_bf16(a, bfr[nt], acc[nt]);
  }
#pragma unroll
  for (int nt = 0; nt < 8; ++nt) {
    int c = nt * 16 + nn;
    float bb = b1[c];
#pragma unroll
    for (int r = 0; r < 8; ++r)
      xs[(r + 8 * half) * DD + c] = acc[nt][r] + bb;
  }

  // --- LayerNorm + ReLU + dot(W2); two lanes per row ---
  float* rowp = &xs[nn * DD + half * 64];
  float s = 0.f, ss = 0.f;
  for (int j = 0; j < 64; ++j) { float v = rowp[j]; s += v; ss += v * v; }
  s  += __shfl_xor(s, 16, 32);
  ss += __shfl_xor(ss, 16, 32);
  float mu  = s * (1.f / 128.f);
  float var = ss * (1.f / 128.f) - mu * mu;
  float rs  = rsqrtf(var + 1e-5f);
  float d = 0.f;
  for (int j = 0; j < 64; ++j) {
    int c = half * 64 + j;
    float v = (rowp[j] - mu) * rs * lng[c] + lnb[c];
    v = fmaxf(v, 0.f);
    d += v * W2[c];
  }
  d += __shfl_xor(d, 16, 32);
  if (lane < 16) out[(bh * NTAR + t0) + nn] = d + b2[0];
}

// ---------------- host launcher ----------------

extern "C" void kernel_launch(void* const* d_in, const int* in_sizes, int n_in,
                              void* d_out, int out_size, void* d_ws, size_t ws_size,
                              hipStream_t stream) {
  (void)in_sizes; (void)n_in; (void)out_size; (void)ws_size;
  const float* A    = (const float*)d_in[0];
  const float* phi  = (const float*)d_in[1];
  const float* Hobs = (const float*)d_in[2];
  const float* Wq   = (const float*)d_in[3];
  const float* bq   = (const float*)d_in[4];
  const float* Wk   = (const float*)d_in[5];
  const float* bk   = (const float*)d_in[6];
  const float* Wv   = (const float*)d_in[7];
  const float* bv   = (const float*)d_in[8];
  const float* W1   = (const float*)d_in[9];
  const float* b1   = (const float*)d_in[10];
  const float* lng  = (const float*)d_in[11];
  const float* lnb  = (const float*)d_in[12];
  const float* W2   = (const float*)d_in[13];
  const float* b2   = (const float*)d_in[14];
  float* out = (float*)d_out;

  char* ws = (char*)d_ws;
  size_t off = 0;
  auto take = [&](size_t bytes) -> char* {
    char* p = ws + off;
    off = (off + bytes + 255) & ~(size_t)255;
    return p;
  };
  unsigned short* Wkb = (unsigned short*)take((size_t)DD * DD * 2);
  unsigned short* Wvb = (unsigned short*)take((size_t)DD * DD * 2);
  unsigned short* W1b = (unsigned short*)take((size_t)DD * DD * 2);
  float*          QA  = (float*)take((size_t)BH * DD * 4);
  float*          QP  = (float*)take((size_t)NTAR * DD * 4);
  unsigned short* Xb  = (unsigned short*)take((size_t)ROWS_KV * DD * 2);
  unsigned short* Kb  = (unsigned short*)take((size_t)ROWS_KV * DD * 2);
  unsigned short* Vtb = (unsigned short*)take((size_t)ROWS_KV * DD * 2);

  cvt_w_kernel<<<(DD * DD + 255) / 256, 256, 0, stream>>>(Wk, Wv, W1, Wkb, Wvb, W1b);
  cvt_x_kernel<<<(ROWS_KV * DD + 255) / 256, 256, 0, stream>>>(Hobs, Xb);
  qa_kernel<<<(BH * DD + 255) / 256, 256, 0, stream>>>(A, Wq, QA);
  qphi_kernel<<<(NTAR * DD + 255) / 256, 256, 0, stream>>>(phi, Wq, bq, QP);
  kv_proj_kernel<<<1024, 32, 0, stream>>>(Xb, Wkb, Wvb, bk, bv, Kb, Vtb);
  attn_ffn_kernel<<<BH * (NTAR / 16), 32, 0, stream>>>(QA, QP, Kb, Vtb, W1b,
                                                       b1, lng, lnb, W2, b2, out);
}
